// GCN3_16552803959363
// MI455X (gfx1250) — matrix operand order repaired and checked by reading, not compiled, not run on hardware
//
#include <hip/hip_runtime.h>
#include <hip/hip_bf16.h>

// ---------------------------------------------------------------------------
// GCN 2-layer forward for MI455X (gfx1250, wave32, WMMA).
//   out = Dd^-1/2 * A * Ds^-1/2 * relu(Dd^-1/2 * A * Ds^-1/2 * X W1 + b1) W2 + b2
// W applied BEFORE aggregation (linear ops commute), so GEMMs run on the
// matrix pipe (v_wmma_f32_16x16x32_f16) and layer-2 scatter moves 64 not 128
// features per edge.
// ---------------------------------------------------------------------------

typedef __attribute__((ext_vector_type(16))) _Float16 v16h;
typedef __attribute__((ext_vector_type(8)))  float    v8f;

#define N_NODES 100000
#define IN_F 128
#define H_F  128
#define OUT_F 64

// ---------------------------------------------------------------------------
__global__ void zero_f32_kernel(float* __restrict__ p, long long n) {
    long long i = (long long)blockIdx.x * blockDim.x + threadIdx.x;
    if (i < n) p[i] = 0.0f;
}

// degree accumulation: one float atomic per endpoint
__global__ void degrees_kernel(const int* __restrict__ src, const int* __restrict__ dst,
                               float* __restrict__ dsrc, float* __restrict__ ddst, int E) {
    int i = blockIdx.x * blockDim.x + threadIdx.x;
    if (i < E) {
        atomicAdd(&dsrc[src[i]], 1.0f);
        atomicAdd(&ddst[dst[i]], 1.0f);
    }
}

// deg -> clamp(deg,1)^-1/2, in place
__global__ void invsqrt_kernel(float* __restrict__ d, int n) {
    int i = blockIdx.x * blockDim.x + threadIdx.x;
    if (i < n) {
        float v = d[i];
        v = fmaxf(v, 1.0f);
        d[i] = rsqrtf(v);
    }
}

// W[K x NC] (row-major f32) -> Wt[NC x K] (row-major f16): transposed so the
// WMMA B-fragment load is two contiguous runs of 8 halfwords per lane.
__global__ void convert_w_t_kernel(const float* __restrict__ W, _Float16* __restrict__ Wt,
                                   int K, int NC) {
    int i = blockIdx.x * blockDim.x + threadIdx.x;
    if (i < K * NC) {
        int k = i / NC, n = i % NC;
        Wt[(size_t)n * K + k] = (_Float16)W[i];
    }
}

// C[M x NC] = A[M x K](f32) @ B[K x NC], B given transposed as Bt[NC x K](f16).
// One wave (blockDim=32) per 16x16 C tile; K consumed 32/step via
// v_wmma_f32_16x16x32_f16, f32 accumulate.
// VGPR fragment layouts per CDNA5 ISA 7.12.2:
//   A 16x32 f16 : lane L holds row M=L%16; elems 0-7 -> K = 8*(L/16)+{0..7},
//                 elems 8-15 -> K = 8*(L/16)+{16..23}
//   B 32x16 f16 : lane L holds col N=L%16; elems 0-7 -> K = 8*(L/16)+{0..7},
//                 elems 8-15 -> K = 8*(L/16)+{16..23}
//   C 16x16 f32 : lane L elem r -> row 8*(L/16)+r, col L%16
template <int K, int NC>
__global__ void gemm_wmma_kernel(const float* __restrict__ A, const _Float16* __restrict__ Bt,
                                 float* __restrict__ C, int M) {
    const int lane  = threadIdx.x & 31;
    const int half  = lane >> 4;          // 0: lanes 0-15, 1: lanes 16-31
    const int l15   = lane & 15;
    const int mtile = blockIdx.x;
    const int ntile = blockIdx.y;

    int row = mtile * 16 + l15;
    int rowc = row < M ? row : (M - 1);   // clamp for safety (M is a multiple of 16 here)
    const int col = ntile * 16 + l15;

    v8f acc = {};
    for (int k0 = 0; k0 < K; k0 += 32) {
        // --- A fragment: two contiguous 32B runs, f32 -> f16 in-register ---
        const float* ap = A + (size_t)rowc * K + k0 + half * 8;
        v16h a;
#pragma unroll
        for (int e = 0; e < 8; ++e) a[e] = (_Float16)ap[e];
#pragma unroll
        for (int e = 0; e < 8; ++e) a[8 + e] = (_Float16)ap[16 + e];
        // --- B fragment: two contiguous 16B runs of f16 per lane ---
        const _Float16* bp = Bt + (size_t)col * K + k0 + half * 8;
        v16h b;
#pragma unroll
        for (int e = 0; e < 8; ++e) b[e] = bp[e];
#pragma unroll
        for (int e = 0; e < 8; ++e) b[8 + e] = bp[16 + e];

        acc = __builtin_amdgcn_wmma_f32_16x16x32_f16(
            /*neg_a=*/false, a, /*neg_b=*/false, b,
            /*c_mod=*/(short)0, acc, /*reuse_a=*/false, /*reuse_b=*/false);
    }

#pragma unroll
    for (int r = 0; r < 8; ++r) {
        int orow = mtile * 16 + half * 8 + r;
        if (orow < M) C[(size_t)orow * NC + col] = acc[r];
    }
}

// Edge SpMM: agg[dst] += X[src] * norm_src[src]. One thread = (edge, 4 feats);
// with F=128 a wave covers one edge -> src/dst scalarize, float4 gathers are
// fully coalesced (512B per edge). X lives in L2 (51MB < 192MB).
template <int F>
__global__ void spmm_kernel(const int* __restrict__ src, const int* __restrict__ dst,
                            const float* __restrict__ X, const float* __restrict__ ns,
                            float* __restrict__ agg, int E) {
    const int CH = F / 4;
    long long gid = (long long)blockIdx.x * blockDim.x + threadIdx.x;
    long long total = (long long)E * CH;
    if (gid >= total) return;
    int e = (int)(gid / CH);
    int c = (int)(gid % CH);
    int s = src[e];
    int d = dst[e];
    float nsv = ns[s];
    const float4 v = *(const float4*)(X + (size_t)s * F + c * 4);
    float* out = agg + (size_t)d * F + c * 4;
    atomicAdd(out + 0, v.x * nsv);
    atomicAdd(out + 1, v.y * nsv);
    atomicAdd(out + 2, v.z * nsv);
    atomicAdd(out + 3, v.w * nsv);
}

// agg = [relu](agg * norm_dst[row] + b[col]), in place
template <bool RELU, int F>
__global__ void finalize_kernel(float* __restrict__ agg, const float* __restrict__ nd,
                                const float* __restrict__ bias, int N) {
    long long gid = (long long)blockIdx.x * blockDim.x + threadIdx.x;
    if (gid >= (long long)N * F) return;
    int i = (int)(gid / F);
    int f = (int)(gid % F);
    float v = agg[gid] * nd[i] + bias[f];
    if (RELU) v = fmaxf(v, 0.0f);
    agg[gid] = v;
}

// ---------------------------------------------------------------------------
extern "C" void kernel_launch(void* const* d_in, const int* in_sizes, int n_in,
                              void* d_out, int out_size, void* d_ws, size_t ws_size,
                              hipStream_t stream) {
    const float* X  = (const float*)d_in[0];   // [N, 128]
    const int*   src = (const int*)d_in[1];    // [E]
    const int*   dst = (const int*)d_in[2];    // [E]
    const float* W1 = (const float*)d_in[3];   // [128, 128]
    const float* b1 = (const float*)d_in[4];   // [128]
    const float* W2 = (const float*)d_in[5];   // [128, 64]
    const float* b2 = (const float*)d_in[6];   // [64]

    const int N = in_sizes[0] / IN_F;          // 100000
    const int E = in_sizes[1];                 // 1600000

    // ---- workspace carving (256B aligned) ----
    size_t off = 0;
    auto carve = [&](size_t bytes) {
        size_t o = off;
        off = (off + bytes + 255) & ~(size_t)255;
        return o;
    };
    char* ws = (char*)d_ws;
    float*     norm_src = (float*)(ws + carve((size_t)N * 4));
    float*     norm_dst = (float*)(ws + carve((size_t)N * 4));
    _Float16*  W1t      = (_Float16*)(ws + carve((size_t)H_F * IN_F * 2));   // [128 x 128]
    _Float16*  W2t      = (_Float16*)(ws + carve((size_t)OUT_F * H_F * 2));  // [64 x 128]
    float*     XW1      = (float*)(ws + carve((size_t)N * H_F * 4));         // [N,128]; reused for HW2 [N,64]
    float*     AGG1     = (float*)(ws + carve((size_t)N * H_F * 4));         // [N,128]; becomes H after finalize
    float*     HW2      = XW1;                                               // reuse
    float*     OUT      = (float*)d_out;                                     // [N,64]

    const int TPB = 256;
    auto blocks = [&](long long n) { return (unsigned)((n + TPB - 1) / TPB); };

    // ---- zero accumulators (harness does not re-zero between replays) ----
    zero_f32_kernel<<<blocks((long long)N), TPB, 0, stream>>>(norm_src, N);
    zero_f32_kernel<<<blocks((long long)N), TPB, 0, stream>>>(norm_dst, N);
    zero_f32_kernel<<<blocks((long long)N * H_F), TPB, 0, stream>>>(AGG1, (long long)N * H_F);
    zero_f32_kernel<<<blocks((long long)N * OUT_F), TPB, 0, stream>>>(OUT, (long long)N * OUT_F);

    // ---- normalization coefficients ----
    degrees_kernel<<<blocks(E), TPB, 0, stream>>>(src, dst, norm_src, norm_dst, E);
    invsqrt_kernel<<<blocks(N), TPB, 0, stream>>>(norm_src, N);
    invsqrt_kernel<<<blocks(N), TPB, 0, stream>>>(norm_dst, N);

    // ---- weights: f32 -> f16, transposed ----
    convert_w_t_kernel<<<blocks(IN_F * H_F), TPB, 0, stream>>>(W1, W1t, IN_F, H_F);
    convert_w_t_kernel<<<blocks(H_F * OUT_F), TPB, 0, stream>>>(W2, W2t, H_F, OUT_F);

    // ---- layer 1: XW1 = X @ W1 (WMMA), spmm, relu(agg*nd + b1) ----
    {
        dim3 grid((N + 15) / 16, H_F / 16);
        gemm_wmma_kernel<IN_F, H_F><<<grid, 32, 0, stream>>>(X, W1t, XW1, N);
    }
    spmm_kernel<H_F><<<blocks((long long)E * (H_F / 4)), TPB, 0, stream>>>(
        src, dst, XW1, norm_src, AGG1, E);
    finalize_kernel<true, H_F><<<blocks((long long)N * H_F), TPB, 0, stream>>>(
        AGG1, norm_dst, b1, N);

    // ---- layer 2: HW2 = H @ W2 (WMMA), spmm into OUT, agg*nd + b2 ----
    {
        dim3 grid((N + 15) / 16, OUT_F / 16);
        gemm_wmma_kernel<H_F, OUT_F><<<grid, 32, 0, stream>>>(AGG1, W2t, HW2, N);
    }
    spmm_kernel<OUT_F><<<blocks((long long)E * (OUT_F / 4)), TPB, 0, stream>>>(
        src, dst, HW2, norm_src, OUT, E);
    finalize_kernel<false, OUT_F><<<blocks((long long)N * OUT_F), TPB, 0, stream>>>(
        OUT, norm_dst, b2, N);
}
